// MLP_tcnn_41927470743708
// MI455X (gfx1250) — matrix operand order repaired and checked by reading, not compiled, not run on hardware
//
#include <hip/hip_runtime.h>

// MI455X / gfx1250 fully-fused MLP: f16 WMMA (16x16x32) with f32 accumulation.
//
// Fast path (needs 162 KB of d_ws):
//   prepass kernels convert fp32 weights -> f16, transposed [N][K+pad], into d_ws.
//   Main kernel: 256 threads = 8 wave32; each wave owns 32 rows (two 16-row
//   A-fragment sets sharing every B fragment), block owns 256 rows.
//   Weights staged per-layer into LDS with pure 16B copies (no cvt).
//   x loads / out stores are non-temporal (streaming, zero reuse); weights stay
//   regular-temporal so the 162KB f16 weight image lives in L2.
// Fallback path: single-kernel version with in-block fp32->f16 staging.

typedef __attribute__((ext_vector_type(16))) _Float16 v16h;
typedef __attribute__((ext_vector_type(8)))  _Float16 v8h;
typedef __attribute__((ext_vector_type(8)))  float    v8f;
typedef __attribute__((ext_vector_type(4)))  float    v4f;

#define THREADS 256

__device__ __forceinline__ v16h make_v16h(v8h lo, v8h hi) {
  v16h r;
#pragma unroll
  for (int i = 0; i < 8; ++i) { r[i] = lo[i]; r[i + 8] = hi[i]; }
  return r;
}

// ---------------- shared fragment helpers ----------------

// B fragment (32x16 f16) for n-tile t, k-tile kt from transposed LDS weights.
// Lane L holds column N=L&15; halves 0..7 -> K = kt*32 + h + 8*(L>>4), halves 8..15 -> K = kt*32 + 16 + (h-8) + 8*(L>>4).
template <int KP>
__device__ __forceinline__ v16h load_bfrag(const _Float16* sW, int t, int kt,
                                           int lane) {
  int n  = t * 16 + (lane & 15);
  int hi = lane >> 4;
  const _Float16* p = sW + n * KP + kt * 32 + hi * 8;
  v8h lo = *(const v8h*)(p);
  v8h hh = *(const v8h*)(p + 16);
  return make_v16h(lo, hh);
}

// A fragment (16x32 f16) from per-wave scratch S[16][40] (one 32-wide K slice).
// Lane L row M=L&15; halves 0..7 -> K=h+8*hi, halves 8..15 -> K=16+(h-8)+8*hi.
__device__ __forceinline__ v16h load_afrag_scratch(const _Float16* S, int lane) {
  int m  = lane & 15;
  int hi = lane >> 4;
  const _Float16* p = S + m * 40;
  v8h lo = *(const v8h*)(p + 8 * hi);
  v8h hh = *(const v8h*)(p + 16 + 8 * hi);
  return make_v16h(lo, hh);
}

// Store one 16x16 f32 D tile (ReLU'd, cvt f16) into scratch K-cols tp*16..+15.
// D layout: VGPR r holds element (M = r + 8*(lane>>4), N = lane&15).
// fmaxf(v,0) == IEEE maximumNumber -> single v_max_num_f32.
__device__ __forceinline__ void store_dtile_relu(_Float16* S, v8f c, int tp,
                                                 int lane) {
  int m0  = (lane >> 4) * 8;
  int col = (lane & 15) + tp * 16;
#pragma unroll
  for (int r = 0; r < 8; ++r) {
    float v = fmaxf(c[r], 0.0f);
    S[(m0 + r) * 40 + col] = (_Float16)v;
  }
}

// Layer-0 A fragment straight from global x (row-major, 64 floats/row).
// Non-temporal: x is streamed exactly once.
__device__ __forceinline__ v16h load_afrag_x(const float* xrow, int k0, int hi) {
  const v4f* p1 = (const v4f*)(xrow + k0 + 8 * hi);
  const v4f* p2 = (const v4f*)(xrow + k0 + 16 + 8 * hi);
  v4f a = __builtin_nontemporal_load(p1);
  v4f b = __builtin_nontemporal_load(p1 + 1);
  v4f c = __builtin_nontemporal_load(p2);
  v4f d = __builtin_nontemporal_load(p2 + 1);
  v16h f;
#pragma unroll
  for (int i = 0; i < 4; ++i) {
    f[i]      = (_Float16)a[i];
    f[4 + i]  = (_Float16)b[i];
    f[8 + i]  = (_Float16)c[i];
    f[12 + i] = (_Float16)d[i];
  }
  return f;
}

// ---------------- fast path: pre-converted f16 weights in d_ws ----------------

// Prepass: convert fp32 W[K][N] -> f16 transposed wt[n*KP + k] in global ws.
template <int KDIM, int NDIM, int KP>
__global__ __launch_bounds__(THREADS) void convert_weights(
    const float* __restrict__ W, _Float16* __restrict__ wt) {
  int e = blockIdx.x * THREADS + threadIdx.x;
  if (e < KDIM * NDIM) {
    int k = e / NDIM;
    int n = e - k * NDIM;
    wt[n * KP + k] = (_Float16)W[e];
  }
}

// Stage pre-converted f16 weights global -> LDS as pure 16B copies.
__device__ __forceinline__ void stage_copy(const _Float16* __restrict__ src,
                                           _Float16* dst, int nchunks) {
  const uint4* s = (const uint4*)src;
  uint4* d = (uint4*)dst;
  for (int i = threadIdx.x; i < nchunks; i += THREADS) d[i] = s[i];
}

// One dense 128-out layer with ReLU for TWO 16-row sets sharing B fragments.
template <int KT, int KP>
__device__ __forceinline__ void dense_layer_relu2(v16h* A0, v16h* A1,
                                                  const _Float16* sW,
                                                  _Float16* S0, _Float16* S1,
                                                  int lane) {
  v16h n0[4], n1[4];
#pragma unroll
  for (int j = 0; j < 4; ++j) {  // pair of 16-wide n-tiles -> one 32-wide K slice
#pragma unroll
    for (int tp = 0; tp < 2; ++tp) {
      v8f a0 = {}, a1 = {};
#pragma unroll
      for (int kt = 0; kt < KT; ++kt) {
        v16h b = load_bfrag<KP>(sW, 2 * j + tp, kt, lane);
        a0 = __builtin_amdgcn_wmma_f32_16x16x32_f16(false, A0[kt], false, b,
                                                    (short)0, a0, false, false);
        a1 = __builtin_amdgcn_wmma_f32_16x16x32_f16(false, A1[kt], false, b,
                                                    (short)0, a1, false, false);
      }
      store_dtile_relu(S0, a0, tp, lane);
      store_dtile_relu(S1, a1, tp, lane);
    }
    n0[j] = load_afrag_scratch(S0, lane);  // same-wave DS ops are in-order
    n1[j] = load_afrag_scratch(S1, lane);
  }
#pragma unroll
  for (int j = 0; j < 4; ++j) { A0[j] = n0[j]; A1[j] = n1[j]; }
}

// ws layout (halves): L0 @0 (128*72), L1..L4 @9216+l*17408 (128*136), L5 @78848 (16*136)
__global__ __launch_bounds__(THREADS) void mlp_fused_f16w(
    const float* __restrict__ x, const _Float16* __restrict__ wf,
    const float* __restrict__ bias, float* __restrict__ out, int nrows) {
  __shared__ __align__(16) _Float16 sW[128 * 136];       // 34,816 B
  __shared__ __align__(16) _Float16 sS[8 * 2 * 16 * 40]; // 20,480 B

  const int lane = threadIdx.x & 31;
  const int wave = threadIdx.x >> 5;
  _Float16* S0 = sS + wave * 2 * 640;
  _Float16* S1 = S0 + 640;

  const int row0 = blockIdx.x * 256 + wave * 32;

  // ---- Layer-0 input: two A-fragment sets (rows row0.. and row0+16..) ----
  v16h A0[4], A1[4];
  A0[2] = A0[3] = A1[2] = A1[3] = v16h{};
  {
    int hi = lane >> 4;
    int r0 = row0 + (lane & 15);       if (r0 >= nrows) r0 = nrows - 1;
    int r1 = row0 + 16 + (lane & 15);  if (r1 >= nrows) r1 = nrows - 1;
    const float* x0 = x + (size_t)r0 * 64;
    const float* x1 = x + (size_t)r1 * 64;
    A0[0] = load_afrag_x(x0, 0, hi);  A0[1] = load_afrag_x(x0, 32, hi);
    A1[0] = load_afrag_x(x1, 0, hi);  A1[1] = load_afrag_x(x1, 32, hi);
  }

  // ---- Layer 0: 64 -> 128, ReLU ----
  stage_copy(wf + 0, sW, 1152);            // 128*72 halves
  __syncthreads();
  dense_layer_relu2<2, 72>(A0, A1, sW, S0, S1, lane);

  // ---- Layers 1..4: 128 -> 128, ReLU ----
  constexpr int offs[4] = {9216, 26624, 44032, 61440};
#pragma unroll
  for (int l = 0; l < 4; ++l) {
    __syncthreads();                       // all waves done reading previous sW
    stage_copy(wf + offs[l], sW, 2176);    // 128*136 halves
    __syncthreads();
    dense_layer_relu2<4, 136>(A0, A1, sW, S0, S1, lane);
  }

  // ---- Layer 5: 128 -> 16, no activation, + bias ----
  __syncthreads();
  stage_copy(wf + 78848, sW, 272);         // 16*136 halves
  __syncthreads();

  v8f c0 = {}, c1 = {};
#pragma unroll
  for (int kt = 0; kt < 4; ++kt) {
    v16h b = load_bfrag<136>(sW, 0, kt, lane);
    c0 = __builtin_amdgcn_wmma_f32_16x16x32_f16(false, A0[kt], false, b,
                                                (short)0, c0, false, false);
    c1 = __builtin_amdgcn_wmma_f32_16x16x32_f16(false, A1[kt], false, b,
                                                (short)0, c1, false, false);
  }

  const float bv = bias[lane & 15];
  const int m0  = (lane >> 4) * 8;
  const int col = lane & 15;
#pragma unroll
  for (int r = 0; r < 8; ++r) {
    int rr = row0 + m0 + r;
    if (rr < nrows)
      __builtin_nontemporal_store(c0[r] + bv, &out[(size_t)rr * 16 + col]);
  }
#pragma unroll
  for (int r = 0; r < 8; ++r) {
    int rr = row0 + 16 + m0 + r;
    if (rr < nrows)
      __builtin_nontemporal_store(c1[r] + bv, &out[(size_t)rr * 16 + col]);
  }
}

// ---------------- fallback path: in-block fp32 staging ----------------

template <int KDIM, int NDIM, int KP>
__device__ __forceinline__ void stage_weights(const float* __restrict__ W,
                                              _Float16* sW) {
#pragma unroll 4
  for (int e = threadIdx.x; e < KDIM * NDIM; e += THREADS) {
    int k = e / NDIM;
    int n = e - k * NDIM;
    sW[n * KP + k] = (_Float16)W[e];
  }
}

template <int KT, int KP>
__device__ __forceinline__ void dense_layer_relu(v16h* A, const _Float16* sW,
                                                 _Float16* S, int lane) {
  v16h nA[4];
#pragma unroll
  for (int j = 0; j < 4; ++j) {
#pragma unroll
    for (int tp = 0; tp < 2; ++tp) {
      v8f acc = {};
#pragma unroll
      for (int kt = 0; kt < KT; ++kt) {
        v16h b = load_bfrag<KP>(sW, 2 * j + tp, kt, lane);
        acc = __builtin_amdgcn_wmma_f32_16x16x32_f16(
            false, A[kt], false, b, (short)0, acc, false, false);
      }
      store_dtile_relu(S, acc, tp, lane);
    }
    nA[j] = load_afrag_scratch(S, lane);
  }
#pragma unroll
  for (int j = 0; j < 4; ++j) A[j] = nA[j];
}

__global__ __launch_bounds__(THREADS) void mlp_fused_fp32w(
    const float* __restrict__ x,  const float* __restrict__ W0,
    const float* __restrict__ W1, const float* __restrict__ W2,
    const float* __restrict__ W3, const float* __restrict__ W4,
    const float* __restrict__ W5, const float* __restrict__ bias,
    float* __restrict__ out, int nrows) {
  __shared__ __align__(16) _Float16 sW[128 * 136];
  __shared__ __align__(16) _Float16 sS[8 * 16 * 40];

  const int lane = threadIdx.x & 31;
  const int wave = threadIdx.x >> 5;
  _Float16* S = sS + wave * 16 * 40;
  const int row0 = blockIdx.x * 128 + wave * 16;

  v16h A[4];
  A[2] = A[3] = v16h{};
  {
    int r = row0 + (lane & 15);
    if (r >= nrows) r = nrows - 1;
    const float* xrow = x + (size_t)r * 64;
    int hi = lane >> 4;
    A[0] = load_afrag_x(xrow, 0, hi);
    A[1] = load_afrag_x(xrow, 32, hi);
  }

  stage_weights<64, 128, 72>(W0, sW);
  __syncthreads();
  dense_layer_relu<2, 72>(A, sW, S, lane);

  const float* const Wmid[4] = {W1, W2, W3, W4};
#pragma unroll
  for (int l = 0; l < 4; ++l) {
    __syncthreads();
    stage_weights<128, 128, 136>(Wmid[l], sW);
    __syncthreads();
    dense_layer_relu<4, 136>(A, sW, S, lane);
  }

  __syncthreads();
  stage_weights<128, 16, 136>(W5, sW);
  __syncthreads();

  v8f acc = {};
#pragma unroll
  for (int kt = 0; kt < 4; ++kt) {
    v16h b = load_bfrag<136>(sW, 0, kt, lane);
    acc = __builtin_amdgcn_wmma_f32_16x16x32_f16(
        false, A[kt], false, b, (short)0, acc, false, false);
  }

  const float bv = bias[lane & 15];
  const int m0  = (lane >> 4) * 8;
  const int col = lane & 15;
#pragma unroll
  for (int r = 0; r < 8; ++r) {
    int rr = row0 + m0 + r;
    if (rr < nrows)
      __builtin_nontemporal_store(acc[r] + bv, &out[(size_t)rr * 16 + col]);
  }
}

// ---------------- host launch ----------------

extern "C" void kernel_launch(void* const* d_in, const int* in_sizes, int n_in,
                              void* d_out, int out_size, void* d_ws,
                              size_t ws_size, hipStream_t stream) {
  const float* x    = (const float*)d_in[0];
  const float* W0   = (const float*)d_in[1];
  const float* W1   = (const float*)d_in[2];
  const float* W2   = (const float*)d_in[3];
  const float* W3   = (const float*)d_in[4];
  const float* W4   = (const float*)d_in[5];
  const float* W5   = (const float*)d_in[6];
  const float* bias = (const float*)d_in[7];
  float* out = (float*)d_out;

  const int nrows = in_sizes[0] / 64;  // N = 524288

  const size_t WS_REQ = (size_t)81024 * sizeof(_Float16);  // 162,048 B
  if (ws_size >= WS_REQ) {
    _Float16* wf = (_Float16*)d_ws;
    convert_weights<64, 128, 72><<<32, THREADS, 0, stream>>>(W0, wf + 0);
    convert_weights<128, 128, 136><<<64, THREADS, 0, stream>>>(W1, wf + 9216);
    convert_weights<128, 128, 136><<<64, THREADS, 0, stream>>>(W2, wf + 26624);
    convert_weights<128, 128, 136><<<64, THREADS, 0, stream>>>(W3, wf + 44032);
    convert_weights<128, 128, 136><<<64, THREADS, 0, stream>>>(W4, wf + 61440);
    convert_weights<128, 16, 136><<<8, THREADS, 0, stream>>>(W5, wf + 78848);

    const int blocks = (nrows + 255) / 256;  // 2048 blocks of 256 rows
    mlp_fused_f16w<<<blocks, THREADS, 0, stream>>>(x, wf, bias, out, nrows);
  } else {
    const int blocks = (nrows + 127) / 128;
    mlp_fused_fp32w<<<blocks, THREADS, 0, stream>>>(x, W0, W1, W2, W3, W4, W5,
                                                    bias, out, nrows);
  }
}
